// NCDEBranch_59803124629643
// MI455X (gfx1250) — matrix operand order repaired and checked by reading, not compiled, not run on hardware
//
#include <hip/hip_runtime.h>

#define CC 32     // input channels
#define HH 128    // hidden dim
#define MM 256    // MLP hidden
#define BB 128    // batch
#define LL 200    // intervals
#define WW 201    // grid times
#define HC 4096   // H*C

#define NTHREADS 512   // 16 waves, 4/SIMD

typedef __attribute__((ext_vector_type(16))) __bf16 v16bf;
typedef __attribute__((ext_vector_type(8)))  float  v8f;
typedef __attribute__((ext_vector_type(4)))  unsigned int v4u;

typedef __attribute__((address_space(1))) const v4u* gv4u_ptr;  // true global loads

union FragBF { v16bf v; v4u q[2]; };

// Opaque pointer anchor: volatile asm is never hoisted, so addresses derived from
// the result are re-formed inside the scan loop each g-eval. The loads themselves
// stay ordinary (cacheable, clause-able) loads -- unlike `volatile` loads, which
// lowered to serialized SCOPE_SYS flat loads.
__device__ __forceinline__ const __bf16* opaque(const __bf16* p) {
  asm volatile("" : "+s"(p));
  return p;
}

// A-matrix 16x32 bf16 fragment: lane m=l&15; hi-half (+16 K offset).
// element e -> K = k0 + hi*16 + e (matches B fragment order)
__device__ __forceinline__ v16bf load_a_frag(const __bf16* base, int m, int K, int k0, int hi) {
  FragBF f;
  const __bf16* p = base + m * K + k0 + hi * 16;
  f.q[0] = *(const v4u*)(p);        // 8 bf16, 16B  (LDS: ds_load_b128)
  f.q[1] = *(const v4u*)(p + 8);    // 8 bf16, 16B
  return f.v;
}

// B-matrix 32x16 bf16 fragment: lane n=l&15; lanes 0-15 K=0..15, lanes 16-31 K=16..31,
// ascending pairs per VGPR -> element e holds K = k0 + hi*16 + e. WT is N-major (row = n, stride K).
// Addresses cast to address_space(1) so these lower to global_load_b128 (LOADcnt only),
// keeping the global weight stream decoupled from the LDS DScnt stream.
__device__ __forceinline__ v16bf load_b_frag(const __bf16* WT, int n, int K, int k0, int hi) {
  FragBF f;
  const __bf16* p = WT + n * K + k0 + hi * 16;
  gv4u_ptr g = (gv4u_ptr)(unsigned long long)p;
  f.q[0] = g[0];                    // 16 contiguous bf16 = 32B per lane
  f.q[1] = g[1];
  return f.v;
}

__device__ __forceinline__ float fast_tanh(float x) {
#if __has_builtin(__builtin_amdgcn_tanhf)
  return __builtin_amdgcn_tanhf(x);              // v_tanh_f32: 1 TRANS op
#elif __has_builtin(__builtin_amdgcn_tanh_f32)
  return __builtin_amdgcn_tanh_f32(x);
#else
  float ax = fabsf(x);
  float t  = __expf(-2.0f * ax);                 // v_exp_f32
  float r  = __fdividef(1.0f - t, 1.0f + t);
  return __builtin_copysignf(r, x);
#endif
}

// One evaluation of g(z_stage, d): h = relu(aZ@W1 + b1); f = tanh(h@W2 + b2);
// kOut[m][h] = sum_c f[m][h*32+c] * d[m][c].  16 waves cooperate; ends with barrier.
__device__ __forceinline__ void geval(
    const __bf16* aZ, __bf16* hS,
    const __bf16* W1T, const __bf16* W2T,
    const float* b1S, const float* __restrict__ b2,
    const float* dS, float* kOut,
    int wv, int hi, int l15)
{
  const __bf16* W1Tl = opaque(W1T);   // anchor weight streams inside the loop
  const __bf16* W2Tl = opaque(W2T);

  // ---- GEMM1: (16xHH) @ (HHxMM) -> hS; all 16 waves, one 16-col tile each ----
  {
    v8f acc1 = {};
    const int n = wv * 16 + l15;
    #pragma unroll
    for (int kk = 0; kk < HH / 32; ++kk) {
      v16bf a = load_a_frag(aZ, l15, HH, kk * 32, hi);
      v16bf b = load_b_frag(W1Tl, n, HH, kk * 32, hi);
      acc1 = __builtin_amdgcn_wmma_f32_16x16x32_bf16(false, a, false, b,
                                                     (short)0, acc1, false, false);
    }
    const float bias = b1S[n];
    #pragma unroll
    for (int v = 0; v < 8; ++v) {
      float x = acc1[v] + bias;
      hS[(v + hi * 8) * MM + n] = (__bf16)(x > 0.f ? x : 0.f);
    }
  }
  __syncthreads();

  // ---- GEMM2: (16xMM) @ (MMxHC); wave owns 256 cols = 16 tiles, 4 passes of 4
  //      (acc = 32 VGPRs; kk unrolled x2 keeps the load window bounded) ----
  const int nbase = wv * 256;
  #pragma unroll
  for (int pass = 0; pass < 4; ++pass) {
    const int n0p = nbase + pass * 64;
    v8f acc[4] = {};
    #pragma unroll 2
    for (int kk = 0; kk < MM / 32; ++kk) {
      v16bf a = load_a_frag(hS, l15, MM, kk * 32, hi);   // reused across 4 tiles
      #pragma unroll
      for (int t = 0; t < 4; ++t) {
        v16bf b = load_b_frag(W2Tl, n0p + t * 16 + l15, MM, kk * 32, hi);
        acc[t] = __builtin_amdgcn_wmma_f32_16x16x32_bf16(false, a, false, b,
                                                         (short)0, acc[t], false, false);
      }
    }

    // epilogue: tanh, multiply by d, reduce over C=32 directly on acc regs
    float dreg[2][8];
    #pragma unroll
    for (int p = 0; p < 2; ++p)
      #pragma unroll
      for (int v = 0; v < 8; ++v)
        dreg[p][v] = dS[(v + hi * 8) * CC + p * 16 + l15];   // c = p*16 + l15

    float sum[2][8];
    #pragma unroll
    for (int j = 0; j < 2; ++j)
      #pragma unroll
      for (int v = 0; v < 8; ++v) sum[j][v] = 0.f;

    #pragma unroll
    for (int t = 0; t < 4; ++t) {
      const float bias2 = b2[n0p + t * 16 + l15];
      const int p = t & 1;
      #pragma unroll
      for (int v = 0; v < 8; ++v) {
        float f = fast_tanh(acc[t][v] + bias2);
        sum[t >> 1][v] = fmaf(f, dreg[p][v], sum[t >> 1][v]);
      }
    }
    #pragma unroll
    for (int j = 0; j < 2; ++j) {
      #pragma unroll
      for (int v = 0; v < 8; ++v) {
        float s = sum[j][v];
        s += __shfl_xor(s, 1, 32);
        s += __shfl_xor(s, 2, 32);
        s += __shfl_xor(s, 4, 32);
        s += __shfl_xor(s, 8, 32);               // lanes 0 / 16 hold the half-sums
        if (l15 == 0) kOut[(v + hi * 8) * HH + (wv * 8 + pass * 2 + j)] = s;
      }
    }
  }
  __syncthreads();
}

// Pre-transpose weights to bf16, N-major, so B-fragments are contiguous 32B/lane.
extern "C" __global__ void ncde_prep(const float* __restrict__ W1,
                                     const float* __restrict__ W2,
                                     __bf16* __restrict__ W1T,
                                     __bf16* __restrict__ W2T)
{
  int i = blockIdx.x * blockDim.x + threadIdx.x;
  if (i < MM * HH) {                       // W1T[n][k] = W1[k][n], n<256, k<128
    int n = i >> 7, k = i & 127;
    W1T[i] = (__bf16)W1[k * MM + n];
  }
  if (i < HC * MM) {                       // W2T[n][k] = W2[k][n], n<4096, k<256
    int n = i >> 8, k = i & 255;
    W2T[i] = (__bf16)W2[k * HC + n];
  }
}

// Persistent scan: 8 blocks x 512 threads; block = 16 batch rows on one WGP.
extern "C" __global__ void __launch_bounds__(NTHREADS, 1)
ncde_scan(const float* __restrict__ coeffs, const float* __restrict__ b1,
          const float* __restrict__ b2, const float* __restrict__ Wz,
          const float* __restrict__ bz, const float* __restrict__ gamma,
          const float* __restrict__ beta, const __bf16* __restrict__ W1T,
          const __bf16* __restrict__ W2T, float* __restrict__ out)
{
  __shared__ __attribute__((aligned(16))) float  zS[16 * HH];    // f32 ODE state
  __shared__ __attribute__((aligned(16))) __bf16 aZ[16 * HH];    // bf16 stage input
  __shared__ __attribute__((aligned(16))) __bf16 hS[16 * MM];    // bf16 hidden
  __shared__ __attribute__((aligned(16))) float  kS[4][16 * HH]; // k1..k4
  __shared__ float dA[16 * CC], dB[16 * CC], d13S[16 * CC], d23S[16 * CC];
  __shared__ float b1S[MM];                      // bias tables cached in LDS
  __shared__ float gS[HH], btS[HH];

  const int tid  = threadIdx.x;
  const int lane = tid & 31;
  const int wv   = tid >> 5;
  const int hi   = lane >> 4;
  const int l15  = lane & 15;
  const int brow0 = blockIdx.x * 16;

  float* aS = kS[3];                             // reuse kS[3] as init staging (a[:,0])

  // init: a[:,0], d_grid[0] = b[:,0], bias tables
  {
    int r = tid >> 5, c = tid & 31;              // tid < 512 == NTHREADS
    size_t base = ((size_t)(brow0 + r) * LL) * (4 * CC);
    aS[tid] = coeffs[base + c];
    dB[tid] = coeffs[base + CC + c];
  }
  if (tid < MM) b1S[tid] = b1[tid];
  if (tid < HH) { gS[tid] = gamma[tid]; btS[tid] = beta[tid]; }
  __syncthreads();
  // z0 = a0 @ Wz + bz (tiny: plain FMAs)
  for (int idx = tid; idx < 16 * HH; idx += NTHREADS) {
    int r = idx >> 7, h = idx & 127;
    float s = bz[h];
    #pragma unroll
    for (int c = 0; c < CC; ++c) s = fmaf(aS[r * CC + c], Wz[c * HH + h], s);
    zS[idx] = s;
  }
  __syncthreads();

  for (int w = 0; w <= LL; ++w) {
    const float* d0 = (w & 1) ? dA : dB;          // deriv at grid time w
    float*       d1 = (w & 1) ? dB : dA;          // deriv at grid time w+1 (written below)

    for (int idx = tid; idx < 16 * HH; idx += NTHREADS) aZ[idx] = (__bf16)zS[idx];
    if (w < LL) {                                 // spline derivs for interval w
      int r = tid >> 5, c = tid & 31;
      size_t base = ((size_t)(brow0 + r) * LL + w) * (4 * CC);
      float bv = coeffs[base + CC + c];
      float tc = coeffs[base + 2 * CC + c];
      float td = coeffs[base + 3 * CC + c];
      d13S[tid] = bv + tc * (1.f / 3.f) + td * (1.f / 9.f);
      d23S[tid] = bv + tc * (2.f / 3.f) + td * (4.f / 9.f);
      d1[tid]   = bv + tc + td;
    }
    __syncthreads();

    // k1 = g(z, d0) — this IS the output field value at grid time w
    geval(aZ, hS, W1T, W2T, b1S, b2, d0, kS[0], wv, hi, l15);

    // LayerNorm(k1) -> out[b][w][:]  (wave r handles batch row r)
    {
      int r = wv;
      float x[4]; float s = 0.f, sq = 0.f;
      #pragma unroll
      for (int u = 0; u < 4; ++u) {
        x[u] = kS[0][r * HH + lane * 4 + u]; s += x[u]; sq += x[u] * x[u];
      }
      #pragma unroll
      for (int m = 1; m < 32; m <<= 1) { s += __shfl_xor(s, m, 32); sq += __shfl_xor(sq, m, 32); }
      float mean = s * (1.f / HH);
      float inv  = rsqrtf(sq * (1.f / HH) - mean * mean + 1e-5f);
      size_t ob = ((size_t)(brow0 + r) * WW + w) * HH;
      #pragma unroll
      for (int u = 0; u < 4; ++u) {
        int h = lane * 4 + u;
        float y = (x[u] - mean) * inv * gS[h] + btS[h];
        out[ob + h] = fminf(fmaxf(y, -1000.f), 1000.f);
      }
    }
    if (w == LL) break;

    // k2 = g(z + k1/3, d13)
    for (int idx = tid; idx < 16 * HH; idx += NTHREADS)
      aZ[idx] = (__bf16)(zS[idx] + kS[0][idx] * (1.f / 3.f));
    __syncthreads();
    geval(aZ, hS, W1T, W2T, b1S, b2, d13S, kS[1], wv, hi, l15);

    // k3 = g(z + k2 - k1/3, d23)
    for (int idx = tid; idx < 16 * HH; idx += NTHREADS)
      aZ[idx] = (__bf16)(zS[idx] + kS[1][idx] - kS[0][idx] * (1.f / 3.f));
    __syncthreads();
    geval(aZ, hS, W1T, W2T, b1S, b2, d23S, kS[2], wv, hi, l15);

    // k4 = g(z + k1 - k2 + k3, d1)
    for (int idx = tid; idx < 16 * HH; idx += NTHREADS)
      aZ[idx] = (__bf16)(zS[idx] + kS[0][idx] - kS[1][idx] + kS[2][idx]);
    __syncthreads();
    geval(aZ, hS, W1T, W2T, b1S, b2, d1, kS[3], wv, hi, l15);

    // z += (k1 + 3(k2+k3) + k4)/8
    for (int idx = tid; idx < 16 * HH; idx += NTHREADS)
      zS[idx] += (kS[0][idx] + 3.f * (kS[1][idx] + kS[2][idx]) + kS[3][idx]) * 0.125f;
    __syncthreads();
  }
}

extern "C" void kernel_launch(void* const* d_in, const int* in_sizes, int n_in,
                              void* d_out, int out_size, void* d_ws, size_t ws_size,
                              hipStream_t stream) {
  (void)in_sizes; (void)n_in; (void)out_size; (void)ws_size;
  const float* coeffs = (const float*)d_in[0];
  const float* W1     = (const float*)d_in[1];
  const float* b1     = (const float*)d_in[2];
  const float* W2     = (const float*)d_in[3];
  const float* b2     = (const float*)d_in[4];
  const float* Wz     = (const float*)d_in[5];
  const float* bz     = (const float*)d_in[6];
  const float* gamma  = (const float*)d_in[7];
  const float* beta   = (const float*)d_in[8];

  __bf16* W1T = (__bf16*)d_ws;          // 256*128 bf16
  __bf16* W2T = W1T + MM * HH;          // 4096*256 bf16
  float*  out = (float*)d_out;

  ncde_prep<<<(HC * MM + 255) / 256, 256, 0, stream>>>(W1, W2, W1T, W2T);
  ncde_scan<<<BB / 16, NTHREADS, 0, stream>>>(coeffs, b1, b2, Wz, bz, gamma, beta,
                                              W1T, W2T, out);
}
